// SplitConv4Pim_group_10445360464132
// MI455X (gfx1250) — matrix operand order repaired and checked by reading, not compiled, not run on hardware
//
#include <hip/hip_runtime.h>

typedef __attribute__((ext_vector_type(16))) _Float16 v16h;
typedef __attribute__((ext_vector_type(8)))  _Float16 v8h;
typedef __attribute__((ext_vector_type(8)))  float    v8f;
typedef int v4i_ __attribute__((vector_size(16)));   // b128 unit for async copies

#define ICC     256
#define OCC     256
#define NGRP    4
#define HWD     56
#define SPATIAL 3136                 // 56*56
#define NBATCH  16
#define NTAP    9

#define MT 128                       // oc tile per workgroup
#define NT 128                       // spatial tile per workgroup
#define KT 32                        // ic chunk = WMMA K
#define N_TILES 25                   // ceil(3136/128)

// X halo tile in LDS: input rows [oh_min-1, oh_min+4], cols [-1,56], 32 ic
#define HALO_R 6
#define HALO_C 58
#define HALO_CELLS (HALO_R * HALO_C)           // 348
#define B_PITCH 40                             // halves per cell: 32 + 8 pad (80B)
#define HALO_LDS   (HALO_CELLS * B_PITCH)      // 13920 halves = 27840 B per buffer
#define N_XFER (HALO_CELLS * 4)                // 1392 16B lane-transfers per stage

// wqf fragment-native layout (halves):
//   off = ((((g*9 + tap)*16 + ocb)*8 + icc)*32 + lane)*16
#define WQF_TOTAL (NGRP * NTAP * 16 * 8 * 32 * 16)   // 2,359,296 halves = 4.5 MB
#define WQF_BYTES ((size_t)WQF_TOTAL * 2)
#define ZPAD_HALVES 256                               // absorbs +ic0+sub*8
#define XH_TOTAL ((size_t)NBATCH * SPATIAL * ICC)     // 12,845,056 halves

#if defined(__HIP_DEVICE_COMPILE__) && \
    __has_builtin(__builtin_amdgcn_global_load_async_to_lds_b128) && \
    __has_builtin(__builtin_amdgcn_s_wait_asynccnt)
#define USE_ASYNC 1
#else
#define USE_ASYNC 0
#endif

// ---------------------------------------------------------------------------
// Pre-pass 1: LSQ quantize + bit-slice weights into WMMA-A-fragment layout.
//   w_q = (((clip(round(w/s),-128,127)+128)>>6 & 3)<<6 - 128) * s
// A-fragment half h of lane l maps to k = (h&7) + (h>>3)*16 + (l>>4)*8.
// ---------------------------------------------------------------------------
__global__ __launch_bounds__(256)
void quant_weights_kernel(const float* __restrict__ w,
                          const float* __restrict__ w_s,
                          _Float16* __restrict__ wqf) {
    int idx = blockIdx.x * 256 + threadIdx.x;
    if (idx >= WQF_TOTAL) return;
    int h    = idx & 15;
    int lane = (idx >> 4) & 31;
    int icc  = (idx >> 9) & 7;
    int ocb  = (idx >> 12) & 15;
    int gt   = idx >> 16;
    int tap  = gt % NTAP;
    int g    = gt / NTAP;

    int k  = (h & 7) + ((h >> 3) << 4) + ((lane >> 4) << 3);
    int ic = (icc << 5) + k;
    int oc = (ocb << 4) + (lane & 15);

    float s   = w_s[icc];
    float val = w[((size_t)(g * OCC + oc) * ICC + ic) * NTAP + tap];
    float wi  = fminf(fmaxf(rintf(val / s), -128.f), 127.f);
    int   u   = (int)wi + 128;
    int   spl = ((u >> 6) & 3) << 6;
    wqf[idx]  = (_Float16)((float)(spl - 128) * s);
}

// ---------------------------------------------------------------------------
// Pre-pass 2: x (f32, [b][ic][sp]) -> xh (f16, [b][sp][ic]) + zero pad region.
// ---------------------------------------------------------------------------
__global__ __launch_bounds__(256)
void xh_convert_kernel(const float* __restrict__ x,
                       _Float16* __restrict__ zpad,
                       _Float16* __restrict__ xh) {
    size_t idx = (size_t)blockIdx.x * 256 + threadIdx.x;
    if (idx < ZPAD_HALVES) zpad[idx] = (_Float16)0.f;
    if (idx >= XH_TOTAL) return;
    int ic = (int)(idx & 255);
    int sp = (int)((idx >> 8) % SPATIAL);
    int b  = (int)(idx / ((size_t)SPATIAL * ICC));
    xh[idx] = (_Float16)x[((size_t)(b * ICC + ic)) * SPATIAL + sp];
}

// ---------------------------------------------------------------------------
// Main kernel: implicit-GEMM conv (9 shifted GEMMs), f16 WMMA, f32 accum.
// 256 threads = 8 waves (wave32), tiled 2(M) x 4(N); each wave: 4x2 accums.
// Double-buffered halo, one barrier per K-step. Staging is 6x16B per thread,
// via async global->LDS copies (ASYNCcnt) when available.
// ---------------------------------------------------------------------------
__global__ __launch_bounds__(256)
void splitconv_wmma_kernel(const _Float16* __restrict__ xh,
                           const _Float16* __restrict__ zpad,
                           const _Float16* __restrict__ wqf,
                           const float* __restrict__ ps_s,
                           float* __restrict__ out) {
    __shared__ _Float16 lds_b[2 * HALO_LDS];    // 55,680 B

    const int tid  = threadIdx.x;
    const int lane = tid & 31;
    const int ln16 = lane & 15;
    const int lhi  = lane >> 4;
    const int wave = tid >> 5;
    const int wm   = wave & 1;              // M band: rows wm*64..
    const int wn   = wave >> 1;             // N band: cols wn*32..

    const int bi      = blockIdx.x / N_TILES;
    const int n_base  = (blockIdx.x % N_TILES) * NT;
    const int oc_base = blockIdx.y * MT;
    const int oh_min  = n_base / HWD;

    // per-lane B-fragment bases (column position in halo) for ni = 0,1
    int lane_base[2];
#pragma unroll
    for (int ni = 0; ni < 2; ++ni) {
        int n   = n_base + wn * 32 + ni * 16 + ln16;
        int ohn = n / HWD;
        int own = n - ohn * HWD;
        lane_base[ni] = ((ohn - oh_min) * HALO_C + own) * B_PITCH;
    }

    const _Float16* xh_b = xh + (size_t)bi * SPATIAL * ICC;

    // ---- precompute the 6 halo transfers owned by this thread ----
    const _Float16* sptr[6];    // source (xh cell run, or zero pad); +ic0 later
    int             doff[6];    // LDS destination offset (halves)
#pragma unroll
    for (int e = 0; e < 6; ++e) {
        int idx  = tid + e * 256;
        int cell = (idx >> 2) < HALO_CELLS ? (idx >> 2) : 0;
        int sub  = idx & 3;
        int r    = cell / HALO_C;
        int c    = cell - r * HALO_C;
        int ih   = oh_min - 1 + r;
        int iw   = c - 1;
        bool valid = ((unsigned)ih < HWD) && ((unsigned)iw < HWD);
        sptr[e] = valid ? (xh_b + (size_t)(ih * HWD + iw) * ICC + sub * 8)
                        : (zpad + sub * 8);
        doff[e] = cell * B_PITCH + sub * 8;
    }

    auto stage = [&](int bufsel, int ic0) {
        _Float16* dst = lds_b + bufsel * HALO_LDS;
#pragma unroll
        for (int e = 0; e < 6; ++e) {
            if (e < 5 || tid < (N_XFER - 5 * 256)) {
#if USE_ASYNC
                __builtin_amdgcn_global_load_async_to_lds_b128(
                    (v4i_*)(sptr[e] + ic0),
                    (v4i_*)(dst + doff[e]),
                    0, 0);
#else
                *(v8h*)(dst + doff[e]) = *(const v8h*)(sptr[e] + ic0);
#endif
            }
        }
    };

    v8f ysum[4][2] = {};

    stage(0, 0);                    // prologue: (g=0, icc=0)
#if USE_ASYNC
    __builtin_amdgcn_s_wait_asynccnt(0);
#endif
    int p = 0;

    for (int g = 0; g < NGRP; ++g) {
        v8f acc[4][2] = {};

        for (int icc = 0; icc < 8; ++icc) {
            __syncthreads();        // buf p staged; buf p^1 free

            const bool last = (g == NGRP - 1) && (icc == 7);
            if (!last) {
                stage(p ^ 1, ((icc + 1) & 7) * KT);
                // hint the chunk after next toward L2 (global_prefetch_b8)
                __builtin_prefetch(sptr[0] + ((icc + 2) & 7) * KT, 0, 1);
            }

            const _Float16* buf = lds_b + p * HALO_LDS;

#pragma unroll
            for (int tap = 0; tap < NTAP; ++tap) {
                const int kh = tap / 3, kw = tap - kh * 3;
                const int tapoff = (kh * HALO_C + kw) * B_PITCH;

                v16h bfrag[2];
#pragma unroll
                for (int ni = 0; ni < 2; ++ni) {
                    const _Float16* bp = buf + lane_base[ni] + tapoff + lhi * 8;
                    ((v8h*)&bfrag[ni])[0] = *(const v8h*)(bp);       // K 8h..8h+7
                    ((v8h*)&bfrag[ni])[1] = *(const v8h*)(bp + 16);  // K 16+8h..23+8h
                }
#pragma unroll
                for (int mi = 0; mi < 4; ++mi) {
                    const int ocb = (oc_base >> 4) + wm * 4 + mi;
                    const size_t off =
                        (size_t)((((g * NTAP + tap) * 16 + ocb) * 8 + icc) * 32
                                 + lane) * 16;
                    v16h afrag = *(const v16h*)(wqf + off);   // coalesced 32B/lane
#pragma unroll
                    for (int ni = 0; ni < 2; ++ni) {
                        acc[mi][ni] = __builtin_amdgcn_wmma_f32_16x16x32_f16(
                            false, afrag, false, bfrag[ni],
                            (short)0, acc[mi][ni], false, false);
                    }
                }
            }
#if USE_ASYNC
            __builtin_amdgcn_s_wait_asynccnt(0);   // drain before next barrier
#endif
            p ^= 1;
        }

        // ---- per-group psum quantization, accumulate ----
        const float sp = ps_s[g];
#pragma unroll
        for (int mi = 0; mi < 4; ++mi)
#pragma unroll
            for (int ni = 0; ni < 2; ++ni)
#pragma unroll
                for (int e = 0; e < 8; ++e) {
                    float q = fminf(fmaxf(rintf(acc[mi][ni][e] / sp),
                                          -128.f), 127.f) * sp;
                    ysum[mi][ni][e] += q;
                }
    }

    // ---- store: D layout => row m = mi*16 + lhi*8 + e, col = ln16 ----
#pragma unroll
    for (int mi = 0; mi < 4; ++mi)
#pragma unroll
        for (int ni = 0; ni < 2; ++ni) {
            const int ncol = n_base + wn * 32 + ni * 16 + ln16;
            if (ncol >= SPATIAL) continue;
#pragma unroll
            for (int e = 0; e < 8; ++e) {
                const int oc = oc_base + wm * 64 + mi * 16 + lhi * 8 + e;
                out[((size_t)bi * OCC + oc) * SPATIAL + ncol] = ysum[mi][ni][e];
            }
        }
}

// ---------------------------------------------------------------------------
extern "C" void kernel_launch(void* const* d_in, const int* in_sizes, int n_in,
                              void* d_out, int out_size, void* d_ws, size_t ws_size,
                              hipStream_t stream) {
    const float* x      = (const float*)d_in[0];   // [16,256,56,56]
    const float* weight = (const float*)d_in[1];   // [1024,256,3,3]
    const float* w_s    = (const float*)d_in[2];   // [8]
    const float* ps_s   = (const float*)d_in[3];   // [4]
    float* out          = (float*)d_out;           // [16,256,56,56]

    // workspace layout: [wqf 4.5MB][zpad 512B][xh 25.7MB]  (~30.2 MB)
    _Float16* wqf  = (_Float16*)d_ws;
    _Float16* zpad = (_Float16*)((char*)d_ws + WQF_BYTES);
    _Float16* xh   = zpad + ZPAD_HALVES;

    quant_weights_kernel<<<(WQF_TOTAL + 255) / 256, 256, 0, stream>>>(weight, w_s, wqf);

    xh_convert_kernel<<<(int)((XH_TOTAL + 255) / 256), 256, 0, stream>>>(x, zpad, xh);

    dim3 grid(NBATCH * N_TILES, OCC / MT, 1);
    splitconv_wmma_kernel<<<grid, 256, 0, stream>>>(xh, zpad, wqf, ps_s, out);
}
